// Encoder_77592879169656
// MI455X (gfx1250) — matrix operand order repaired and checked by reading, not compiled, not run on hardware
//
#include <hip/hip_runtime.h>
#include <stdint.h>

#define NN    20000
#define EE    80000
#define DIMV  32
#define NGRP  128
#define KEDGE 128          // edge-MLP hidden
#define KTOT  4096         // fused GEMM K = 32*128

typedef __attribute__((ext_vector_type(16))) _Float16 v16h;
typedef __attribute__((ext_vector_type(8)))  _Float16 v8h;
typedef __attribute__((ext_vector_type(8)))  float    v8f;

__device__ __forceinline__ float sigmoidf_(float x) { return 1.0f / (1.0f + __expf(-x)); }

// ---------------- lin0: out = relu(x @ W^T + b); h0 = out ----------------
__global__ void k_lin0(const float* __restrict__ x, const float* __restrict__ w,
                       const float* __restrict__ b, float* __restrict__ xfeat,
                       float* __restrict__ h0) {
    int tid = blockIdx.x * blockDim.x + threadIdx.x;     // N*32 exact
    int n = tid >> 5, d = tid & 31;
    const float* xr = x + (size_t)n * 32;
    const float* wr = w + (size_t)d * 32;
    float v = b[d];
#pragma unroll
    for (int k = 0; k < 32; ++k) v += xr[k] * wr[k];
    v = fmaxf(v, 0.0f);
    xfeat[tid] = v; h0[tid] = v;
}

// ---------------- in-degree counts ----------------
__global__ void k_count(const int* __restrict__ ei, float* __restrict__ cnt) {
    int e = blockIdx.x * blockDim.x + threadIdx.x;
    if (e < EE) atomicAdd(&cnt[ei[EE + e]], 1.0f);
}

// ---------------- edge MLP layer1: hE = relu(ea @ w1^T + b1)  (f16) ----------------
__global__ void k_edge_mlp(const float* __restrict__ ea, const float* __restrict__ w1,
                           const float* __restrict__ b1, _Float16* __restrict__ hE) {
    int tid = blockIdx.x * blockDim.x + threadIdx.x;     // E*128 exact
    int e = tid >> 7, kk = tid & 127;
    const float* ar = ea + (size_t)e * 5;
    const float* wr = w1 + (size_t)kk * 5;
    float v = b1[kk];
#pragma unroll
    for (int j = 0; j < 5; ++j) v += ar[j] * wr[j];
    hE[tid] = (_Float16)fmaxf(v, 0.0f);
}

// ---------------- pack w2 into WMMA B-fragment layout ----------------
// frag f = t*2+nt (t=kstep 0..127, nt=ntile 0..1); lane L; elem j (0..15)
// B[Klocal, n] with Klocal = 16*(L>>4)+j, n = nt*16 + (L&15)
// kglob = t*32+Klocal; i = kglob>>7; k = kglob&127; value = w2[(i*32 + o)*128 + k]
__global__ void k_bpack(const float* __restrict__ w2, _Float16* __restrict__ bp) {
    int tid = blockIdx.x * blockDim.x + threadIdx.x;     // 128*2*32*16 = 131072 exact
    int j  = tid & 15;
    int L  = (tid >> 4) & 31;
    int f  = tid >> 9;
    int nt = f & 1;
    int t  = f >> 1;
    int Klocal = ((L >> 4) << 4) + j;
    int kglob  = t * 32 + Klocal;
    int i = kglob >> 7;
    int k = kglob & 127;
    int o = nt * 16 + (L & 15);
    bp[tid] = (_Float16)w2[(size_t)(i * 32 + o) * 128 + k];
}

// ---------------- fused NNConv message GEMM + scatter-add ----------------
// msg[e,o] = sum_{i,k} x_src[e,i]*h[e,k]*w2[i*32+o,k]  + sum_i x_src[e,i]*b2[i*32+o]
// One wave per 16-edge tile; K-loop over 128 steps of 32; 2 WMMA per step.
__global__ __launch_bounds__(128)
void k_msg_gemm(const float* __restrict__ xfeat, const _Float16* __restrict__ hE,
                const _Float16* __restrict__ bpack, const float* __restrict__ b2,
                const int* __restrict__ ei, float* __restrict__ sacc) {
    __shared__ __align__(32) _Float16 hT[4][16][KEDGE];
    __shared__ __align__(32) _Float16 xT[4][16][DIMV];
    __shared__ float b2s[DIMV * DIMV];

    const int tid  = threadIdx.x;          // 0..127
    const int w    = tid >> 5;
    const int lane = tid & 31;
    const long tile = (long)blockIdx.x * 4 + w;
    const long eg0  = tile * 16;

    for (int q = tid; q < DIMV * DIMV; q += 128) b2s[q] = b2[q];

    for (int q = lane; q < 16 * KEDGE; q += 32) {
        int r = q >> 7, c = q & 127;
        long e = eg0 + r; if (e > EE - 1) e = EE - 1;
        hT[w][r][c] = hE[e * KEDGE + c];
    }
    for (int q = lane; q < 16 * DIMV; q += 32) {
        int r = q >> 5, c = q & 31;
        long e = eg0 + r; if (e > EE - 1) e = EE - 1;
        int s = ei[e];                               // src node
        xT[w][r][c] = (_Float16)xfeat[(size_t)s * 32 + c];
    }
    __syncthreads();

    v8f acc0, acc1;
#pragma unroll
    for (int r = 0; r < 8; ++r) { acc0[r] = 0.0f; acc1[r] = 0.0f; }

    const int M     = lane & 15;
    const int base0 = (lane >> 4) * 16;
    const _Float16* hrow = &hT[w][M][0];
    const _Float16* fb0  = bpack + (size_t)lane * 16;

    for (int t = 0; t < 128; ++t) {
        const int i    = t >> 2;
        const int hoff = ((t & 3) << 5) + base0;
        _Float16 sx = xT[w][M][i];
        v8h sv;
#pragma unroll
        for (int j = 0; j < 8; ++j) sv[j] = sx;
        v8h g1 = *(const v8h*)(hrow + hoff);
        v8h g2 = *(const v8h*)(hrow + hoff + 8);
        g1 = g1 * sv; g2 = g2 * sv;
        v16h a = __builtin_shufflevector(g1, g2, 0,1,2,3,4,5,6,7,8,9,10,11,12,13,14,15);

        const _Float16* fb = fb0 + (size_t)t * 1024;
        v16h b0 = *(const v16h*)fb;
        v16h b1 = *(const v16h*)(fb + 512);
        if (t + 1 < 128) __builtin_prefetch(fb + 1024, 0, 1);

        acc0 = __builtin_amdgcn_wmma_f32_16x16x32_f16(false, a, false, b0,
                                                      (short)0, acc0, false, false);
        acc1 = __builtin_amdgcn_wmma_f32_16x16x32_f16(false, a, false, b1,
                                                      (short)0, acc1, false, false);
    }

    // epilogue: b2 bias term + scatter to destination accumulator
#pragma unroll
    for (int r = 0; r < 8; ++r) {
        const int Mr = r + ((lane >> 4) << 3);
        const int o  = lane & 15;
        float bias0 = 0.0f, bias1 = 0.0f;
#pragma unroll
        for (int i2 = 0; i2 < 32; ++i2) {
            float xv = (float)xT[w][Mr][i2];
            bias0 += xv * b2s[i2 * 32 + o];
            bias1 += xv * b2s[i2 * 32 + 16 + o];
        }
        long e = eg0 + Mr;
        if (e < EE) {
            int d = ei[EE + e];                      // dst node
            atomicAdd(&sacc[(size_t)d * 32 + o],      acc0[r] + bias0);
            atomicAdd(&sacc[(size_t)d * 32 + o + 16], acc1[r] + bias1);
        }
    }
}

// ---------------- segment mean + conv bias + relu ----------------
__global__ void k_finalize(const float* __restrict__ sacc, const float* __restrict__ cnt,
                           const float* __restrict__ bias, float* __restrict__ out) {
    int tid = blockIdx.x * blockDim.x + threadIdx.x;     // N*32 exact
    int n = tid >> 5, o = tid & 31;
    float v = sacc[tid] / fmaxf(cnt[n], 1.0f) + bias[o];
    out[tid] = fmaxf(v, 0.0f);
}

// ---------------- GRU cell ----------------
__global__ void k_gru(const float* __restrict__ m, const float* __restrict__ hin,
                      const float* __restrict__ wih, const float* __restrict__ whh,
                      const float* __restrict__ bih, const float* __restrict__ bhh,
                      float* __restrict__ hout, float* __restrict__ xfeat) {
    int tid = blockIdx.x * blockDim.x + threadIdx.x;     // N*32 exact
    int n = tid >> 5, j = tid & 31;
    const float* mr = m + (size_t)n * 32;
    const float* hr = hin + (size_t)n * 32;
    float gir = bih[j], giz = bih[32 + j], gin = bih[64 + j];
    float ghr = bhh[j], ghz = bhh[32 + j], ghn = bhh[64 + j];
#pragma unroll
    for (int k = 0; k < 32; ++k) {
        float mv = mr[k], hv = hr[k];
        gir += mv * wih[(size_t)j * 32 + k];
        giz += mv * wih[(size_t)(32 + j) * 32 + k];
        gin += mv * wih[(size_t)(64 + j) * 32 + k];
        ghr += hv * whh[(size_t)j * 32 + k];
        ghz += hv * whh[(size_t)(32 + j) * 32 + k];
        ghn += hv * whh[(size_t)(64 + j) * 32 + k];
    }
    float r = sigmoidf_(gir + ghr);
    float z = sigmoidf_(giz + ghz);
    float nn = tanhf(gin + r * ghn);
    float hv = (1.0f - z) * nn + z * hr[j];
    hout[tid] = hv; xfeat[tid] = hv;
}

// ---------------- group bounds over sorted batch ----------------
__global__ void k_gb_init(int* __restrict__ gs, int* __restrict__ ge) {
    int g = threadIdx.x;
    if (g < NGRP) { gs[g] = NN; ge[g] = 0; }
}
__global__ void k_gb(const int* __restrict__ batch, int* __restrict__ gs, int* __restrict__ ge) {
    int n = blockIdx.x * blockDim.x + threadIdx.x;
    if (n < NN) {
        int b = batch[n];
        atomicMin(&gs[b], n);
        atomicMax(&ge[b], n + 1);
    }
}

// ---------------- Set2Set LSTM cell ----------------
__global__ void k_lstm(const float* __restrict__ qstar, const float* __restrict__ hin,
                       const float* __restrict__ cin, const float* __restrict__ wih,
                       const float* __restrict__ whh, const float* __restrict__ bih,
                       const float* __restrict__ bhh, float* __restrict__ hout,
                       float* __restrict__ cout) {
    int tid = blockIdx.x * blockDim.x + threadIdx.x;     // 128*32 exact
    int g = tid >> 5, u = tid & 31;
    const float* q  = qstar + (size_t)g * 64;
    const float* hr = hin + (size_t)g * 32;
    float gi = bih[u] + bhh[u];
    float gf = bih[32 + u] + bhh[32 + u];
    float gg = bih[64 + u] + bhh[64 + u];
    float go = bih[96 + u] + bhh[96 + u];
#pragma unroll
    for (int k = 0; k < 64; ++k) {
        float qv = q[k];
        gi += qv * wih[(size_t)u * 64 + k];
        gf += qv * wih[(size_t)(32 + u) * 64 + k];
        gg += qv * wih[(size_t)(64 + u) * 64 + k];
        go += qv * wih[(size_t)(96 + u) * 64 + k];
    }
#pragma unroll
    for (int k = 0; k < 32; ++k) {
        float hv = hr[k];
        gi += hv * whh[(size_t)u * 32 + k];
        gf += hv * whh[(size_t)(32 + u) * 32 + k];
        gg += hv * whh[(size_t)(64 + u) * 32 + k];
        go += hv * whh[(size_t)(96 + u) * 32 + k];
    }
    float c = sigmoidf_(gf) * cin[tid] + sigmoidf_(gi) * tanhf(gg);
    float h = sigmoidf_(go) * tanhf(c);
    cout[tid] = c; hout[tid] = h;
}

// ---------------- Set2Set attention: segment softmax + weighted sum ----------------
__global__ __launch_bounds__(256)
void k_attn(const float* __restrict__ xg, const float* __restrict__ h,
            const int* __restrict__ gs, const int* __restrict__ ge,
            float* __restrict__ qstar) {
    const int g = blockIdx.x;
    const int tid = threadIdx.x;
    __shared__ float qv[32];
    __shared__ float red[256];
    __shared__ float racc[32];
    if (tid < 32) { qv[tid] = h[(size_t)g * 32 + tid]; racc[tid] = 0.0f; }
    __syncthreads();
    const int s = gs[g], e = ge[g];

    float lmax = -3.4e38f;
    for (int n = s + tid; n < e; n += 256) {
        const float* xr = xg + (size_t)n * 32;
        float d = 0.0f;
#pragma unroll
        for (int k = 0; k < 32; ++k) d += xr[k] * qv[k];
        lmax = fmaxf(lmax, d);
    }
    red[tid] = lmax; __syncthreads();
    for (int st = 128; st > 0; st >>= 1) {
        if (tid < st) red[tid] = fmaxf(red[tid], red[tid + st]);
        __syncthreads();
    }
    const float emax = red[0]; __syncthreads();

    float lsum = 0.0f;
    for (int n = s + tid; n < e; n += 256) {
        const float* xr = xg + (size_t)n * 32;
        float d = 0.0f;
#pragma unroll
        for (int k = 0; k < 32; ++k) d += xr[k] * qv[k];
        lsum += __expf(d - emax);
    }
    red[tid] = lsum; __syncthreads();
    for (int st = 128; st > 0; st >>= 1) {
        if (tid < st) red[tid] += red[tid + st];
        __syncthreads();
    }
    const float denom = red[0]; __syncthreads();

    for (int n = s + tid; n < e; n += 256) {
        const float* xr = xg + (size_t)n * 32;
        float d = 0.0f;
#pragma unroll
        for (int k = 0; k < 32; ++k) d += xr[k] * qv[k];
        float a = __expf(d - emax);
#pragma unroll
        for (int o = 0; o < 32; ++o) atomicAdd(&racc[o], a * xr[o]);
    }
    __syncthreads();
    if (tid < 32) {
        qstar[(size_t)g * 64 + tid] = qv[tid];
        qstar[(size_t)g * 64 + 32 + tid] = (denom > 0.0f) ? racc[tid] / denom : 0.0f;
    }
}

// ---------------- broadcast grouped outputs back to nodes ----------------
__global__ void k_gather_out(const int* __restrict__ batch, const float* __restrict__ qmu,
                             const float* __restrict__ qlv, float* __restrict__ out) {
    int tid = blockIdx.x * blockDim.x + threadIdx.x;     // N*64 exact
    int n = tid >> 6, j = tid & 63;
    const size_t base = (size_t)2 * NN * 32;
    int g = batch[n];
    out[base + tid] = qmu[(size_t)g * 64 + j];
    out[base + (size_t)NN * 64 + tid] = qlv[(size_t)g * 64 + j];
}

// ===================== host orchestration =====================
extern "C" void kernel_launch(void* const* d_in, const int* in_sizes, int n_in,
                              void* d_out, int out_size, void* d_ws, size_t ws_size,
                              hipStream_t stream) {
    const float* x       = (const float*)d_in[0];
    const float* ea      = (const float*)d_in[1];
    const float* lin0_w  = (const float*)d_in[2];
    const float* lin0_b  = (const float*)d_in[3];
    // conv param blocks: c0..c4 at 4 + 5*p : {w1,b1,w2,b2,bias}
    const float* gru_wih = (const float*)d_in[29];
    const float* gru_whh = (const float*)d_in[30];
    const float* gru_bih = (const float*)d_in[31];
    const float* gru_bhh = (const float*)d_in[32];
    const float* mu_wih  = (const float*)d_in[33];
    const float* mu_whh  = (const float*)d_in[34];
    const float* mu_bih  = (const float*)d_in[35];
    const float* mu_bhh  = (const float*)d_in[36];
    const float* lv_wih  = (const float*)d_in[37];
    const float* lv_whh  = (const float*)d_in[38];
    const float* lv_bih  = (const float*)d_in[39];
    const float* lv_bhh  = (const float*)d_in[40];
    const int*   ei      = (const int*)d_in[41];
    const int*   batch   = (const int*)d_in[42];
    float* out = (float*)d_out;

    // workspace bump allocator (256-byte aligned)
    size_t off = 0;
    char* ws = (char*)d_ws;
    auto alloc = [&](size_t bytes) -> void* {
        void* p = ws + off;
        off = (off + bytes + 255) & ~(size_t)255;
        return p;
    };
    float*    xfeat = (float*)alloc((size_t)NN * 32 * 4);
    float*    hA    = (float*)alloc((size_t)NN * 32 * 4);
    float*    hB    = (float*)alloc((size_t)NN * 32 * 4);
    float*    mbuf  = (float*)alloc((size_t)NN * 32 * 4);
    float*    sacc  = (float*)alloc((size_t)NN * 32 * 4);
    float*    cnt   = (float*)alloc((size_t)NN * 4);
    float*    gmu   = (float*)alloc((size_t)NN * 32 * 4);
    float*    glv   = (float*)alloc((size_t)NN * 32 * 4);
    _Float16* hE    = (_Float16*)alloc((size_t)EE * KEDGE * 2);
    _Float16* bpack = (_Float16*)alloc((size_t)KTOT * 32 * 2);
    float*    qmu   = (float*)alloc((size_t)NGRP * 64 * 4);
    float*    qlv   = (float*)alloc((size_t)NGRP * 64 * 4);
    float*    s2h0  = (float*)alloc((size_t)NGRP * 32 * 4);
    float*    s2h1  = (float*)alloc((size_t)NGRP * 32 * 4);
    float*    s2c0  = (float*)alloc((size_t)NGRP * 32 * 4);
    float*    s2c1  = (float*)alloc((size_t)NGRP * 32 * 4);
    int*      gsb   = (int*)alloc((size_t)NGRP * 4);
    int*      geb   = (int*)alloc((size_t)NGRP * 4);
    (void)ws_size; (void)in_sizes; (void)n_in; (void)out_size;

    const int NTH = 256;
    const int gN32 = (NN * 32) / NTH;           // 2500
    const int gE   = (EE + NTH - 1) / NTH;
    const int gEM  = (EE * KEDGE) / NTH;        // 40000
    const int gBP  = (KTOT * 32) / NTH;         // 512
    const int gGEMM = (EE / 16) / 4;            // 1250 blocks of 4 waves

    auto conv_pass = [&](int pidx, const float* inFeat, float* outBuf, bool prep) {
        const float* w1   = (const float*)d_in[4 + 5 * pidx + 0];
        const float* b1   = (const float*)d_in[4 + 5 * pidx + 1];
        const float* w2   = (const float*)d_in[4 + 5 * pidx + 2];
        const float* b2   = (const float*)d_in[4 + 5 * pidx + 3];
        const float* bias = (const float*)d_in[4 + 5 * pidx + 4];
        if (prep) {
            k_edge_mlp<<<gEM, NTH, 0, stream>>>(ea, w1, b1, hE);
            k_bpack<<<gBP, NTH, 0, stream>>>(w2, bpack);
        }
        hipMemsetAsync(sacc, 0, (size_t)NN * 32 * 4, stream);
        k_msg_gemm<<<gGEMM, 128, 0, stream>>>(inFeat, hE, bpack, b2, ei, sacc);
        k_finalize<<<gN32, NTH, 0, stream>>>(sacc, cnt, bias, outBuf);
    };

    auto set2set = [&](const float* xg, const float* wih, const float* whh,
                       const float* bih, const float* bhh, float* qstar) {
        hipMemsetAsync(qstar, 0, (size_t)NGRP * 64 * 4, stream);
        hipMemsetAsync(s2h0, 0, (size_t)NGRP * 32 * 4, stream);
        hipMemsetAsync(s2c0, 0, (size_t)NGRP * 32 * 4, stream);
        float* hIn = s2h0; float* hOut = s2h1;
        float* cIn = s2c0; float* cOut = s2c1;
        for (int step = 0; step < 3; ++step) {
            k_lstm<<<(NGRP * 32) / NTH, NTH, 0, stream>>>(qstar, hIn, cIn, wih, whh,
                                                          bih, bhh, hOut, cOut);
            k_attn<<<NGRP, 256, 0, stream>>>(xg, hOut, gsb, geb, qstar);
            float* t;
            t = hIn; hIn = hOut; hOut = t;
            t = cIn; cIn = cOut; cOut = t;
        }
    };

    // degree counts + group bounds (input-only, once per launch)
    hipMemsetAsync(cnt, 0, (size_t)NN * 4, stream);
    k_count<<<gE, NTH, 0, stream>>>(ei, cnt);
    k_gb_init<<<1, NGRP, 0, stream>>>(gsb, geb);
    k_gb<<<(NN + NTH - 1) / NTH, NTH, 0, stream>>>(batch, gsb, geb);

    // lin0
    k_lin0<<<gN32, NTH, 0, stream>>>(x, lin0_w, lin0_b, xfeat, hA);

    // 3x (NNConv(c0) + GRU); hE/bpack for c0 hoisted (edge-only)
    float* hCur = hA; float* hNext = hB;
    for (int it = 0; it < 3; ++it) {
        conv_pass(0, xfeat, mbuf, it == 0);
        k_gru<<<gN32, NTH, 0, stream>>>(mbuf, hCur, gru_wih, gru_whh, gru_bih, gru_bhh,
                                        hNext, xfeat);
        float* t = hCur; hCur = hNext; hNext = t;
    }

    // head convs: node_mu, node_lv straight into d_out; gmu/glv into ws
    conv_pass(1, xfeat, out, true);                       // node_mu
    conv_pass(2, xfeat, out + (size_t)NN * 32, true);     // node_lv
    conv_pass(3, xfeat, gmu, true);
    conv_pass(4, xfeat, glv, true);

    // set2set pools
    set2set(gmu, mu_wih, mu_whh, mu_bih, mu_bhh, qmu);
    set2set(glv, lv_wih, lv_whh, lv_bih, lv_bhh, qlv);

    // broadcast grouped outputs to per-node slots
    k_gather_out<<<(NN * 64) / NTH, NTH, 0, stream>>>(batch, qmu, qlv, out);
}
